// Multihead_CrossAttention_Q2V_35485019800105
// MI455X (gfx1250) — matrix operand order repaired and checked by reading, not compiled, not run on hardware
//
#include <hip/hip_runtime.h>
#include <hip/hip_bf16.h>

// MI455X / gfx1250, wave32, WMMA f32_16x16x32_f16.
// Forward math of the reference reduces to plain MHA (gauss/fck branch cancels:
// w2 = weights * rowscalar, renormalized over k => w2 == weights exactly).

typedef _Float16 half8 __attribute__((ext_vector_type(8)));
typedef _Float16 v16h  __attribute__((ext_vector_type(16)));
typedef float    v8f   __attribute__((ext_vector_type(8)));

constexpr int BS = 64, QL = 32, KLEN = 1024, DIM = 1024, NH = 16, DH = 64;

__device__ __forceinline__ v16h cat16(half8 lo, half8 hi) {
  return __builtin_shufflevector(lo, hi, 0,1,2,3,4,5,6,7,8,9,10,11,12,13,14,15);
}

__device__ __forceinline__ half8 cvt8v(float4 f0, float4 f1) {
  half8 h;
  h[0]=(_Float16)f0.x; h[1]=(_Float16)f0.y; h[2]=(_Float16)f0.z; h[3]=(_Float16)f0.w;
  h[4]=(_Float16)f1.x; h[5]=(_Float16)f1.y; h[6]=(_Float16)f1.z; h[7]=(_Float16)f1.w;
  return h;
}

__device__ __forceinline__ v8f wmma_f16(v16h a, v16h b, v8f c) {
  return __builtin_amdgcn_wmma_f32_16x16x32_f16(false, a, false, b, (short)0, c,
                                                false, false);
}

// ---------------------------------------------------------------------------
// C[M,N] = A[M,K] * B[N,K]^T + bias ; A,B row-major K-contiguous.
// MODE 0: f16 row-major out, (c+bias)*scale     (Q/K projections)
// MODE 1: f16 transposed per-batch out[b][n][m%1024] (V projection)
// MODE 2: f32 row-major out + bias              (final output projection)
// AF16:   A source is f16 (context) instead of f32.
// 256 threads = 8 waves (2Mx4N groups); tile 128x128, K-step 32; each wave
// 64x32 C = 4x2 WMMA tiles. Double-buffered staging: next K-slab is fetched
// into registers while WMMAs consume the current LDS tile.
// ---------------------------------------------------------------------------
template <int MODE, bool AF16>
__global__ __launch_bounds__(256) void gemm_kernel(
    const void* __restrict__ Araw, const float* __restrict__ B,
    const float* __restrict__ bias, void* __restrict__ Oraw,
    int M, int N, int K, float scale) {
  __shared__ __align__(16) _Float16 As[128 * 32];
  __shared__ __align__(16) _Float16 Bs[128 * 32];

  const int t    = threadIdx.x;
  const int lane = t & 31, w = t >> 5;
  const int wm   = w >> 2, wn = w & 3;
  const int sel  = lane >> 4, l16 = lane & 15;
  const int mBase = blockIdx.y * 128;
  const int nBase = blockIdx.x * 128;
  const int ldRow = t >> 1;         // 0..127
  const int ldCol = (t & 1) * 16;   // 0 or 16

  float4 ra0, ra1, ra2, ra3, rb0, rb1, rb2, rb3;
  half8  ha0, ha1;

  auto fetch = [&](int k0) {
    if (AF16) {
      const _Float16* src =
          (const _Float16*)Araw + (size_t)(mBase + ldRow) * K + k0 + ldCol;
      ha0 = *(const half8*)src;
      ha1 = *(const half8*)(src + 8);
    } else {
      const float* src =
          (const float*)Araw + (size_t)(mBase + ldRow) * K + k0 + ldCol;
      ra0 = *(const float4*)src;        ra1 = *(const float4*)(src + 4);
      ra2 = *(const float4*)(src + 8);  ra3 = *(const float4*)(src + 12);
    }
    const float* srcB = B + (size_t)(nBase + ldRow) * K + k0 + ldCol;
    rb0 = *(const float4*)srcB;         rb1 = *(const float4*)(srcB + 4);
    rb2 = *(const float4*)(srcB + 8);   rb3 = *(const float4*)(srcB + 12);
  };
  auto stage = [&]() {
    if (AF16) {
      *(half8*)&As[ldRow * 32 + ldCol]     = ha0;
      *(half8*)&As[ldRow * 32 + ldCol + 8] = ha1;
    } else {
      *(half8*)&As[ldRow * 32 + ldCol]     = cvt8v(ra0, ra1);
      *(half8*)&As[ldRow * 32 + ldCol + 8] = cvt8v(ra2, ra3);
    }
    *(half8*)&Bs[ldRow * 32 + ldCol]     = cvt8v(rb0, rb1);
    *(half8*)&Bs[ldRow * 32 + ldCol + 8] = cvt8v(rb2, rb3);
  };

  v8f c[4][2] = {};
  fetch(0);
  stage();
  __syncthreads();

  for (int k0 = 0; k0 < K; k0 += 32) {
    const bool more = (k0 + 32) < K;
    if (more) fetch(k0 + 32);          // overlap global loads with WMMAs below
    for (int mi = 0; mi < 4; ++mi) {
      const int lrow = wm * 64 + mi * 16 + l16;
      v16h a = cat16(*(const half8*)&As[lrow * 32 + sel * 8],
                     *(const half8*)&As[lrow * 32 + 16 + sel * 8]);
      for (int ni = 0; ni < 2; ++ni) {
        const int lnr = wn * 32 + ni * 16 + l16;
        v16h bb = cat16(*(const half8*)&Bs[lnr * 32 + sel * 8],
                        *(const half8*)&Bs[lnr * 32 + 16 + sel * 8]);
        c[mi][ni] = wmma_f16(a, bb, c[mi][ni]);
      }
    }
    __syncthreads();
    if (more) { stage(); __syncthreads(); }
  }

  for (int mi = 0; mi < 4; ++mi) {
    for (int ni = 0; ni < 2; ++ni) {
      const int col  = nBase + wn * 32 + ni * 16 + l16;
      const int row0 = mBase + wm * 64 + mi * 16 + sel * 8;
      const float bv = bias[col];
      if (MODE == 0) {
        _Float16* O = (_Float16*)Oraw;
        for (int r = 0; r < 8; ++r)
          O[(size_t)(row0 + r) * N + col] = (_Float16)((c[mi][ni][r] + bv) * scale);
      } else if (MODE == 1) {
        _Float16* O = (_Float16*)Oraw;
        half8 h;
        for (int r = 0; r < 8; ++r) h[r] = (_Float16)(c[mi][ni][r] + bv);
        // out[b][col][m%1024], b = row0/1024 ; r-rows are memory-contiguous
        size_t addr = (size_t)(row0 >> 10) * ((size_t)N * 1024) +
                      (size_t)col * 1024 + (size_t)(row0 & 1023);
        *(half8*)&O[addr] = h;
      } else {
        float* O = (float*)Oraw;
        for (int r = 0; r < 8; ++r)
          O[(size_t)(row0 + r) * N + col] = c[mi][ni][r] + bv;
      }
    }
  }
}

// ---------------------------------------------------------------------------
// Fused attention per (b,h): scores = Qp_head(32x64) Kp_head(1024x64)^T -> LDS
// (f16, 64 KB), masked; in-place softmax in LDS; context = W(LDS) * VpT^T.
// Weights never touch HBM. 256 threads = 8 waves.
// ---------------------------------------------------------------------------
__global__ __launch_bounds__(256) void fused_attn_kernel(
    const _Float16* __restrict__ Qp, const _Float16* __restrict__ Kp,
    const _Float16* __restrict__ VpT, const unsigned char* __restrict__ mask,
    _Float16* __restrict__ Ctx) {
  __shared__ __align__(16) _Float16 Sl[QL * KLEN];   // 64 KB
  const int h = blockIdx.x, b = blockIdx.y;
  const int t = threadIdx.x, lane = t & 31, w = t >> 5;
  const int sel = lane >> 4, l16 = lane & 15;

  // ---- Phase 1: this wave computes the 32 x 128 strip n0..n0+127 of QK^T
  {
    const int n0 = w * 128;
    v16h a[2][2];                                    // [kstep][mtile]
    for (int ks = 0; ks < 2; ++ks)
      for (int mi = 0; mi < 2; ++mi) {
        const _Float16* src =
            Qp + (size_t)(b * QL + mi * 16 + l16) * DIM + h * DH + ks * 32;
        a[ks][mi] = cat16(*(const half8*)(src + sel * 8),
                          *(const half8*)(src + 16 + sel * 8));
      }
    for (int ni = 0; ni < 8; ++ni) {
      v8f c[2] = {};
      for (int ks = 0; ks < 2; ++ks) {
        const _Float16* src = Kp + (size_t)(b * KLEN + n0 + ni * 16 + l16) * DIM +
                              h * DH + ks * 32;
        v16h bb = cat16(*(const half8*)(src + sel * 8), *(const half8*)(src + 16 + sel * 8));
        for (int mi = 0; mi < 2; ++mi) c[mi] = wmma_f16(a[ks][mi], bb, c[mi]);
      }
      const int col = n0 + ni * 16 + l16;
      const bool ok = mask[b * KLEN + col] != 0;
      for (int mi = 0; mi < 2; ++mi) {
        const int row0 = mi * 16 + sel * 8;
        for (int r = 0; r < 8; ++r) {
          float v = ok ? c[mi][r] : -__builtin_inff();
          Sl[(row0 + r) * KLEN + col] = (_Float16)v;
        }
      }
    }
  }
  __syncthreads();

  // ---- Phase 2: softmax in LDS; 8 threads/row, 128 cols each (disjoint R/W)
  {
    const int row = t >> 3, seg = t & 7;
    _Float16* p = &Sl[row * KLEN + seg * 128];
    float mx = -__builtin_inff();
    for (int i = 0; i < 128; i += 8) {
      half8 hv = *(const half8*)&p[i];
      for (int j = 0; j < 8; ++j) mx = fmaxf(mx, (float)hv[j]);
    }
    for (int off = 1; off < 8; off <<= 1) mx = fmaxf(mx, __shfl_xor(mx, off, 32));
    float sum = 0.f;
    for (int i = 0; i < 128; i += 8) {
      half8 hv = *(const half8*)&p[i];
      for (int j = 0; j < 8; ++j) sum += __expf((float)hv[j] - mx);
    }
    for (int off = 1; off < 8; off <<= 1) sum += __shfl_xor(sum, off, 32);
    const float inv = 1.f / sum;
    for (int i = 0; i < 128; i += 8) {
      half8 hv = *(const half8*)&p[i];
      half8 o;
      for (int j = 0; j < 8; ++j)
        o[j] = (_Float16)(__expf((float)hv[j] - mx) * inv);
      *(half8*)&p[i] = o;
    }
  }
  __syncthreads();

  // ---- Phase 3: context = W(32x1024, LDS) * VpT_head(64x1024)^T
  {
    const int wm = w >> 2, wn = w & 3;
    const int lrow = wm * 16 + l16;
    const size_t vrow = ((size_t)b * DIM + h * DH + wn * 16 + l16) * KLEN;
    v8f c = {};
    for (int k0 = 0; k0 < KLEN; k0 += 32) {
      v16h aa = cat16(*(const half8*)&Sl[lrow * KLEN + k0 + sel * 8],
                      *(const half8*)&Sl[lrow * KLEN + k0 + 16 + sel * 8]);
      v16h bb = cat16(*(const half8*)(VpT + vrow + k0 + sel * 8),
                      *(const half8*)(VpT + vrow + k0 + 16 + sel * 8));
      c = wmma_f16(aa, bb, c);
    }
    for (int r = 0; r < 8; ++r) {
      const int q = wm * 16 + sel * 8 + r;
      const int d = wn * 16 + l16;
      Ctx[((size_t)b * QL + q) * DIM + h * DH + d] = (_Float16)c[r];
    }
  }
}

// ---------------------------------------------------------------------------
extern "C" void kernel_launch(void* const* d_in, const int* in_sizes, int n_in,
                              void* d_out, int out_size, void* d_ws, size_t ws_size,
                              hipStream_t stream) {
  (void)in_sizes; (void)n_in; (void)out_size; (void)ws_size;
  const float* query = (const float*)d_in[0];
  const float* key   = (const float*)d_in[1];
  const float* value = (const float*)d_in[2];
  const unsigned char* mask = (const unsigned char*)d_in[3];
  const float* q_w = (const float*)d_in[4];
  const float* q_b = (const float*)d_in[5];
  const float* k_w = (const float*)d_in[6];
  const float* k_b = (const float*)d_in[7];
  const float* v_w = (const float*)d_in[8];
  const float* v_b = (const float*)d_in[9];
  const float* out_w = (const float*)d_in[10];
  const float* out_b = (const float*)d_in[11];
  // d_in[12..15] (fck_w/b, gauss_w/b) cancel in the forward computation.

  char* ws = (char*)d_ws;
  const size_t MB = 1 << 20;
  _Float16* Qp  = (_Float16*)(ws);              //   4 MB (2048 x 1024 f16)
  _Float16* Kp  = (_Float16*)(ws + 4 * MB);     // 128 MB (65536 x 1024 f16)
  _Float16* VpT = (_Float16*)(ws + 132 * MB);   // 128 MB ([b][1024][1024] f16)
  _Float16* Ctx = (_Float16*)(ws + 260 * MB);   //   4 MB (2048 x 1024 f16)

  const int Mkv = BS * KLEN;   // 65536
  const int Mq  = BS * QL;     // 2048

  // K = key @ k_w^T + k_b                   (f16 out)
  gemm_kernel<0, false><<<dim3(DIM / 128, Mkv / 128), 256, 0, stream>>>(
      key, k_w, k_b, Kp, Mkv, DIM, DIM, 1.0f);
  // V^T = (value @ v_w^T + v_b)^T per batch (f16 out, transposed)
  gemm_kernel<1, false><<<dim3(DIM / 128, Mkv / 128), 256, 0, stream>>>(
      value, v_w, v_b, VpT, Mkv, DIM, DIM, 1.0f);
  // Q = (query @ q_w^T + q_b) / sqrt(64)
  gemm_kernel<0, false><<<dim3(DIM / 128, Mq / 128), 256, 0, stream>>>(
      query, q_w, q_b, Qp, Mq, DIM, DIM, 0.125f);
  // fused scores -> mask -> softmax -> context, per (b,h)
  fused_attn_kernel<<<dim3(NH, BS), 256, 0, stream>>>(Qp, Kp, VpT, mask, Ctx);
  // out = context @ out_w^T + out_b         (f32 out)
  gemm_kernel<2, true><<<dim3(DIM / 128, Mq / 128), 256, 0, stream>>>(
      Ctx, out_w, out_b, d_out, Mq, DIM, DIM, 1.0f);
}
